// GInvariantLSTMLayer_21242908246552
// MI455X (gfx1250) — compile-verified, matrix operand order checked by reading, not run on hardware
//
#include <hip/hip_runtime.h>

// ---------------------------------------------------------------------------
// Types for CDNA5 WMMA (wave32)
// ---------------------------------------------------------------------------
typedef __attribute__((ext_vector_type(16))) __bf16 v16bf;
typedef __attribute__((ext_vector_type(8)))  float  v8f;

#define N_BATCH 128
#define SEQ_LEN 512
#define IN_DIM  512
#define HID     1024

// ---------------------------------------------------------------------------
// f32 -> bf16 round-to-nearest-even
// ---------------------------------------------------------------------------
__device__ __forceinline__ unsigned short f2bf(float x) {
    unsigned int u = __float_as_uint(x);
    u += 0x7FFFu + ((u >> 16) & 1u);
    return (unsigned short)(u >> 16);
}

__device__ __forceinline__ float sigmoidf(float x) {
    return 1.0f / (1.0f + __expf(-x));
}

// ---------------------------------------------------------------------------
// Kernel 1: S[n,l] = sum over 512 features of X[n,l,:]
// One wave32 per row; 16 floats per lane (4x float4), shfl_xor reduction.
// ---------------------------------------------------------------------------
__global__ void __launch_bounds__(256) rowsum512(const float* __restrict__ X,
                                                 float* __restrict__ S) {
    const int lane = threadIdx.x & 31;
    const int row  = blockIdx.x * 8 + (threadIdx.x >> 5);   // 0 .. 65535
    const float4* p = (const float4*)(X + row * IN_DIM + lane * 16);
    float4 a = p[0], b = p[1], c = p[2], d = p[3];
    float v = (a.x + a.y) + (a.z + a.w) + (b.x + b.y) + (b.z + b.w)
            + (c.x + c.y) + (c.z + c.w) + (d.x + d.y) + (d.z + d.w);
#pragma unroll
    for (int m = 16; m >= 1; m >>= 1) v += __shfl_xor(v, m, 32);
    if (lane == 0) S[row] = v;
}

// ---------------------------------------------------------------------------
// Kernel 2: convert W_{i,f,g,o} (f32 [1024][1024], row = out col, col = k)
// to bf16, same layout, concatenated -> wbf[4096][1024].
// Row-major W is already the per-lane-contiguous B-fragment layout.
// ---------------------------------------------------------------------------
__global__ void __launch_bounds__(256) conv_w(const float* __restrict__ Wi,
                                              const float* __restrict__ Wf,
                                              const float* __restrict__ Wg,
                                              const float* __restrict__ Wo,
                                              unsigned short* __restrict__ wbf) {
    const int idx = blockIdx.x * 256 + threadIdx.x;   // 0 .. 4*1024*1024-1
    const int g   = idx >> 20;
    const int r   = idx & 0xFFFFF;
    const float* W = (g == 0) ? Wi : (g == 1) ? Wf : (g == 2) ? Wg : Wo;
    wbf[idx] = f2bf(W[r]);
}

// ---------------------------------------------------------------------------
// Kernel 3: zero initial state: c = 0, h(bf16) = 0
// ---------------------------------------------------------------------------
__global__ void __launch_bounds__(256) init_state(float* __restrict__ c,
                                                  unsigned short* __restrict__ hbf) {
    const int idx = blockIdx.x * 256 + threadIdx.x;   // 0 .. 131071
    c[idx]   = 0.0f;
    hbf[idx] = 0;
}

// ---------------------------------------------------------------------------
// Fragment helpers: two 16B loads per lane (bf16 16x32 A/B lane layout).
// ---------------------------------------------------------------------------
union AB { v16bf v; uint4 q[2]; };
union CF { v8f  v; float f[8]; };

__device__ __forceinline__ void load_frag(AB& f, const unsigned short* p) {
    f.q[0] = *(const uint4*)(p);
    f.q[1] = *(const uint4*)(p + 16);
}

// ---------------------------------------------------------------------------
// Kernel 4: per-step fused gate GEMM.
//   G[128][4096] = h[128][1024](bf16) @ Wall^T   (4 gates fused along N)
// Wave tile: 16(M) x 64(N); K loop via v_wmma_f32_16x16x32_bf16, software-
// pipelined 2-deep so the loads for chunk k+32 are in flight while the WMMAs
// for chunk k execute (removes the s_wait_loadcnt 0x0 before every WMMA).
//
// bf16 A-matrix 16x32 lane layout (ISA 7.12.2):
//   lanes 0-15 : M=lane,    K = {0..7, 16..23}  of chunk
//   lanes 16-31: M=lane-16, K = {8..15, 24..31} of chunk
// B symmetric with N in place of M; since B[k][n] = W[n][k], a lane's column
// is one row of W -> contiguous along k. No transpose needed.
// ---------------------------------------------------------------------------
__global__ void __launch_bounds__(128) lstm_gates_gemm(
        const unsigned short* __restrict__ hbf,   // [128][1024] bf16
        const unsigned short* __restrict__ wbf,   // [4096][1024] bf16
        float* __restrict__ G) {                  // [128][4096] f32
    const int lane = threadIdx.x & 31;
    const int wave = threadIdx.x >> 5;
    const int tile = blockIdx.x * 4 + wave;       // 0 .. 511
    const int mbase = (tile & 7) * 16;            // 8 M-tiles
    const int ct    = tile >> 3;                  // 0 .. 63 column tiles of 64

    const int lr   = lane & 15;
    const int hi   = lane >> 4;
    const int koff = hi * 8;

    const unsigned short* aptr = hbf + (mbase + lr) * HID + koff;
    const unsigned short* bp0  = wbf + (ct * 64 +  0 + lr) * HID + koff;
    const unsigned short* bp1  = bp0 + 16 * HID;
    const unsigned short* bp2  = bp0 + 32 * HID;
    const unsigned short* bp3  = bp0 + 48 * HID;

    CF acc[4];
#pragma unroll
    for (int j = 0; j < 4; ++j)
#pragma unroll
        for (int r = 0; r < 8; ++r) acc[j].f[r] = 0.0f;

    AB a0, a1;
    AB b0[4], b1[4];

    // Prologue: chunk 0 into buffer 0.
    load_frag(a0,    aptr);
    load_frag(b0[0], bp0);
    load_frag(b0[1], bp1);
    load_frag(b0[2], bp2);
    load_frag(b0[3], bp3);

    // 32 K-chunks, processed two per iteration with alternating buffers.
#pragma unroll 4
    for (int k0 = 0; k0 < HID; k0 += 64) {
        const int kA = k0 + 32;                   // always valid (<= 992)
        load_frag(a1,    aptr + kA);
        load_frag(b1[0], bp0 + kA);
        load_frag(b1[1], bp1 + kA);
        load_frag(b1[2], bp2 + kA);
        load_frag(b1[3], bp3 + kA);
        __builtin_prefetch(aptr + k0 + 128, 0, 1);    // global_prefetch_b8
        __builtin_prefetch(bp0  + k0 + 128, 0, 1);

#pragma unroll
        for (int j = 0; j < 4; ++j)
            acc[j].v = __builtin_amdgcn_wmma_f32_16x16x32_bf16(
                false, a0.v, false, b0[j].v, (short)0, acc[j].v, false, false);

        const int kB = k0 + 64;
        if (kB < HID) {
            load_frag(a0,    aptr + kB);
            load_frag(b0[0], bp0 + kB);
            load_frag(b0[1], bp1 + kB);
            load_frag(b0[2], bp2 + kB);
            load_frag(b0[3], bp3 + kB);
        }

#pragma unroll
        for (int j = 0; j < 4; ++j)
            acc[j].v = __builtin_amdgcn_wmma_f32_16x16x32_bf16(
                false, a1.v, false, b1[j].v, (short)0, acc[j].v, false, false);
    }

    // C/D layout: lanes 0-15 -> N=lane, M = vgpr r; lanes 16-31 -> M = r+8.
#pragma unroll
    for (int j = 0; j < 4; ++j) {
        const int col = ct * 64 + j * 16 + lr;
#pragma unroll
        for (int r = 0; r < 8; ++r) {
            const int row = mbase + r + hi * 8;
            G[row * 4096 + col] = acc[j].f[r];
        }
    }
}

// ---------------------------------------------------------------------------
// Kernel 5: per-step pointwise LSTM update.
//   pre_g = S[n,t]*lam_g[h] + binv_g[h];  gate = sigmoid(pre + G + b)
//   c = f*c + i*g;  h = o*sigmoid(c);  write out[n,t,h] and bf16 h.
// ---------------------------------------------------------------------------
__global__ void __launch_bounds__(256) lstm_pointwise(
        const float* __restrict__ S,      // [128][512]
        const float* __restrict__ G,      // [128][4096]
        const float* __restrict__ lam_i, const float* __restrict__ binv_i, const float* __restrict__ b_i,
        const float* __restrict__ lam_f, const float* __restrict__ binv_f, const float* __restrict__ b_f,
        const float* __restrict__ lam_g, const float* __restrict__ binv_g, const float* __restrict__ b_g,
        const float* __restrict__ lam_o, const float* __restrict__ binv_o, const float* __restrict__ b_o,
        float* __restrict__ c, unsigned short* __restrict__ hbf,
        float* __restrict__ out, int t) {
    const int idx = blockIdx.x * 256 + threadIdx.x;   // 0 .. 131071
    const int n = idx >> 10;
    const int h = idx & 1023;
    const float s = S[n * SEQ_LEN + t];
    const float* gn = G + n * 4096;

    const float pi = s * lam_i[h] + binv_i[h] + gn[h]        + b_i[h];
    const float pf = s * lam_f[h] + binv_f[h] + gn[1024 + h] + b_f[h];
    const float pg = s * lam_g[h] + binv_g[h] + gn[2048 + h] + b_g[h];
    const float po = s * lam_o[h] + binv_o[h] + gn[3072 + h] + b_o[h];

    const float gi = sigmoidf(pi);
    const float gf = sigmoidf(pf);
    const float gg = sigmoidf(pg);   // sigmoid (not tanh), as in source
    const float go = sigmoidf(po);

    const float cn = gf * c[idx] + gi * gg;
    c[idx] = cn;
    const float hn = go * sigmoidf(cn);

    hbf[idx] = f2bf(hn);
    out[(n * SEQ_LEN + t) * HID + h] = hn;
}

// ---------------------------------------------------------------------------
// Host-side launch. Workspace layout (bytes):
//   S    @ 0          : 128*512*4        = 262144
//   wbf  @ 262144     : 4*1024*1024*2    = 8388608
//   hbf  @ 8650752    : 128*1024*2       = 262144
//   c    @ 8912896    : 128*1024*4       = 524288
//   G    @ 9437184    : 128*4096*4       = 2097152
//   total ~ 11.0 MB
// ---------------------------------------------------------------------------
extern "C" void kernel_launch(void* const* d_in, const int* in_sizes, int n_in,
                              void* d_out, int out_size, void* d_ws, size_t ws_size,
                              hipStream_t stream) {
    const float* X      = (const float*)d_in[0];
    const float* lam_i  = (const float*)d_in[1];
    const float* binv_i = (const float*)d_in[2];
    const float* W_i    = (const float*)d_in[3];
    const float* b_i    = (const float*)d_in[4];
    const float* lam_f  = (const float*)d_in[5];
    const float* binv_f = (const float*)d_in[6];
    const float* W_f    = (const float*)d_in[7];
    const float* b_f    = (const float*)d_in[8];
    const float* lam_g  = (const float*)d_in[9];
    const float* binv_g = (const float*)d_in[10];
    const float* W_g    = (const float*)d_in[11];
    const float* b_g    = (const float*)d_in[12];
    const float* lam_o  = (const float*)d_in[13];
    const float* binv_o = (const float*)d_in[14];
    const float* W_o    = (const float*)d_in[15];
    const float* b_o    = (const float*)d_in[16];

    char* ws = (char*)d_ws;
    float*          S   = (float*)(ws);
    unsigned short* wbf = (unsigned short*)(ws + 262144);
    unsigned short* hbf = (unsigned short*)(ws + 262144 + 8388608);
    float*          c   = (float*)(ws + 262144 + 8388608 + 262144);
    float*          G   = (float*)(ws + 262144 + 8388608 + 262144 + 524288);
    float* out = (float*)d_out;

    rowsum512 <<<(N_BATCH * SEQ_LEN) / 8, 256, 0, stream>>>(X, S);
    conv_w    <<<(4 * HID * HID) / 256, 256, 0, stream>>>(W_i, W_f, W_g, W_o, wbf);
    init_state<<<(N_BATCH * HID) / 256, 256, 0, stream>>>(c, hbf);

    for (int t = 0; t < SEQ_LEN; ++t) {
        lstm_gates_gemm<<<128, 128, 0, stream>>>(hbf, wbf, G);
        lstm_pointwise <<<(N_BATCH * HID) / 256, 256, 0, stream>>>(
            S, G,
            lam_i, binv_i, b_i, lam_f, binv_f, b_f,
            lam_g, binv_g, b_g, lam_o, binv_o, b_o,
            c, hbf, out, t);
    }
}
